// Skip4Net_9010841387784
// MI455X (gfx1250) — matrix operand order repaired and checked by reading, not compiled, not run on hardware
//
#include <hip/hip_runtime.h>
#include <hip/hip_bf16.h>
#include <math.h>
#include <stdint.h>

// ---------------------------------------------------------------------------
// Types for CDNA5 WMMA / TDM
// ---------------------------------------------------------------------------
typedef __attribute__((ext_vector_type(16))) __bf16 v16bf;
typedef __attribute__((ext_vector_type(8)))  float  v8f;
typedef __attribute__((ext_vector_type(4)))  unsigned int v4u;
typedef __attribute__((ext_vector_type(8)))  int    v8i;
typedef __attribute__((ext_vector_type(4)))  int    v4i;

#define NPIX  262144            // 512*512
#define WDIM  512

#if __has_builtin(__builtin_amdgcn_tensor_load_to_lds) && \
    __has_builtin(__builtin_amdgcn_s_wait_tensorcnt)
#define HAVE_TDM 1
#else
#define HAVE_TDM 0
#endif

// reflect-pad index for pad=1 over [0,511] (jnp 'reflect': -1 -> 1, 512 -> 510)
__device__ __forceinline__ int refl(int v) {
    v = (v < 0) ? -v : v;
    return (v > 511) ? 1022 - v : v;
}

#if HAVE_TDM
// ---------------------------------------------------------------------------
// Issue a 2D TDM load: tile_d1 rows of tile_d0 8-byte elements, row stride
// stride_e (8-byte units), global tile start gaddr -> LDS byte offset lds_off.
// D# bit packing per CDNA5 ISA ch.8 (group0 128b, group1 256b).
// ---------------------------------------------------------------------------
__device__ __forceinline__ void tdm_load_2d(unsigned long long gaddr,
                                            unsigned int lds_off,
                                            unsigned int tile_d0,
                                            unsigned int tile_d1,
                                            unsigned long long stride_e) {
    v4u g0;
    g0.x = 1u;                                    // count=1, user mode, no gather
    g0.y = lds_off;                               // lds_addr (bytes)
    g0.z = (unsigned int)(gaddr & 0xffffffffull); // global_addr[31:0]
    g0.w = (unsigned int)((gaddr >> 32) & 0x01fffffful) | (2u << 30); // [56:32] | type=2

    const unsigned long long td0 = 0x7fffffffull; // huge tensor dims: no OOB clip
    const unsigned long long td1 = 0x7fffffffull;
    v8i g1;
    g1[0] = (int)(3u << 16);                                      // data_size=8B, mask=0
    g1[1] = (int)((td0 & 0xffffu) << 16);                         // tensor_dim0[15:0]
    g1[2] = (int)(((td0 >> 16) & 0xffffu) | ((td1 & 0xffffu) << 16));
    g1[3] = (int)(((td1 >> 16) & 0xffffu) | (tile_d0 << 16));     // | tile_dim0
    g1[4] = (int)(tile_d1 & 0xffffu);                             // tile_dim1, tile_dim2=0
    g1[5] = (int)(stride_e & 0xffffffffull);                      // dim0_stride[31:0]
    g1[6] = (int)((stride_e >> 32) & 0xffffull);                  // dim0_stride[47:32]
    g1[7] = 0;

    v4i gz = {0, 0, 0, 0};
#if __clang_major__ >= 23
    v8i gz8 = {0, 0, 0, 0, 0, 0, 0, 0};
    __builtin_amdgcn_tensor_load_to_lds(g0, g1, gz, gz, gz8, 0);
#else
    __builtin_amdgcn_tensor_load_to_lds(g0, g1, gz, gz, 0);
#endif
}
#endif

// ---------------------------------------------------------------------------
// Weight packing into WMMA A fragments (16-bit A 16x32), contiguous k order (B matches):
//   lane m   (0..15): K = 0..15
//   lane m+16       : K = 16..31
// ---------------------------------------------------------------------------
__global__ void pack_weights_kernel(const float* __restrict__ w1,
                                    const float* __restrict__ w2,
                                    __bf16* __restrict__ wA,
                                    __bf16* __restrict__ wB) {
    int idx = blockIdx.x * blockDim.x + threadIdx.x;
    const int NA = 8 * 9 * 512;
    const int NB = 36 * 512;
    if (idx < NA) {
        int frag = idx >> 9;
        int within = idx & 511;
        int lane = within >> 4;
        int e = within & 15;
        int mtile = frag / 9, tap = frag % 9;
        int m = lane & 15;
        int k = (lane < 16) ? e : e + 16;
        float v = w1[(mtile * 16 + m) * 288 + k * 9 + tap];   // w1 (128,32,3,3)
        wA[idx] = (__bf16)v;
    } else if (idx < NA + NB) {
        int j = idx - NA;
        int frag = j >> 9;
        int within = j & 511;
        int lane = within >> 4;
        int e = within & 15;
        int tap = frag >> 2, cch = frag & 3;
        int m = lane & 15;
        int k = (lane < 16) ? e : e + 16;
        int ic = cch * 32 + k;
        float v = (m < 3) ? w2[m * 1152 + ic * 9 + tap] : 0.0f; // w2 (3,128,3,3)
        wB[j] = (__bf16)v;
    }
}

// ---------------------------------------------------------------------------
// x1 (32,512,512) f32 channel-major -> x1T (512*512, 32) bf16 channel-inner
// ---------------------------------------------------------------------------
__global__ void pack_x1_kernel(const float* __restrict__ x,
                               __bf16* __restrict__ x1T) {
    int p = blockIdx.x * blockDim.x + threadIdx.x;
    if (p >= NPIX) return;
    __bf16* dst = x1T + ((size_t)p << 5);
#pragma unroll
    for (int c = 0; c < 32; ++c)
        dst[c] = (__bf16)x[c * NPIX + p];
}

// ---------------------------------------------------------------------------
// conv1: 3x3, 32->128, reflect pad, leaky_relu(0.2).
// Block = 64-pixel strip. Input patch (3 x 66 x 32 bf16 = 12672 B) staged in
// LDS (TDM for interior blocks, cooperative copy at edges). A fragments held
// in registers across the 4 tiles; two accumulators break WMMA serialization.
// ---------------------------------------------------------------------------
__global__ void __launch_bounds__(256)
conv1_wmma_kernel(const __bf16* __restrict__ x1T,
                  const __bf16* __restrict__ wA,
                  const float*  __restrict__ b1,
                  __bf16* __restrict__ hT) {
    __shared__ __bf16 sbuf[3 * 66 * 32];          // 12672 B
    int wrow = blockIdx.x >> 3;                   // image row (dim2)
    int col0 = (blockIdx.x & 7) << 6;             // strip base column (dim3)
    int wave = threadIdx.x >> 5;                  // out-channel tile
    int lane = threadIdx.x & 31;
    int n = lane & 15;                            // B column = pixel in tile
    int cbase = (lane < 16) ? 0 : 16;             // K half carried by this lane

    bool interior = (wrow > 0) && (wrow < 511) && (col0 > 0) && (col0 < 448);
#if HAVE_TDM
    if (interior) {
        if (threadIdx.x < 32) {
            unsigned long long gaddr = (unsigned long long)(uintptr_t)x1T +
                (((unsigned long long)((wrow - 1) * WDIM + col0 - 1)) << 6);
            unsigned lds_off = (unsigned)(uintptr_t)&sbuf[0];
            // 3 rows x (66*32*2/8 = 528) 8B elems, row pitch 512*32*2/8 = 4096
            tdm_load_2d(gaddr, lds_off, 528u, 3u, 4096ull);
            __builtin_amdgcn_s_wait_tensorcnt((short)0);
        }
        __syncthreads();
    } else
#endif
    {
        for (int t = threadIdx.x; t < 3 * 66; t += 256) {
            int r = t / 66, c = t % 66;
            int rr = refl(wrow + r - 1);
            int cc = refl(col0 + c - 1);
            const uint4* src = (const uint4*)(x1T + (((size_t)(rr * WDIM + cc)) << 5));
            uint4* dst = (uint4*)(sbuf + t * 32);
#pragma unroll
            for (int q = 0; q < 4; ++q) dst[q] = src[q];
        }
        __syncthreads();
    }

    // speculative prefetch of the next strip (emits global_prefetch_b8)
    if (col0 < 448)
        __builtin_prefetch(x1T + ((size_t)(wrow * WDIM + col0 + 64 + n) << 5), 0, 0);

    // A fragments: 9 taps, resident in registers for the whole strip
    v16bf Af[9];
#pragma unroll
    for (int t9 = 0; t9 < 9; ++t9)
        Af[t9] = *(const v16bf*)(wA + (((wave * 9 + t9) << 9) + (lane << 4)));

#pragma unroll
    for (int t = 0; t < 4; ++t) {
        v8f acc0 = {}, acc1 = {};
#pragma unroll
        for (int i = 0; i < 3; ++i) {
#pragma unroll
            for (int j = 0; j < 3; ++j) {
                int tap = i * 3 + j;
                v16bf b = *(const v16bf*)(sbuf + (((i * 66 + t * 16 + n + j)) << 5) + cbase);
                if (tap & 1)
                    acc1 = __builtin_amdgcn_wmma_f32_16x16x32_bf16(
                        false, Af[tap], false, b, (short)0, acc1, false, false);
                else
                    acc0 = __builtin_amdgcn_wmma_f32_16x16x32_bf16(
                        false, Af[tap], false, b, (short)0, acc0, false, false);
            }
        }
        size_t pbase = ((size_t)(wrow * WDIM + col0 + t * 16 + n)) << 7;  // *128
        int mofs = (lane < 16) ? 0 : 8;
#pragma unroll
        for (int r = 0; r < 8; ++r) {
            int ch = wave * 16 + r + mofs;
            float y = acc0[r] + acc1[r] + b1[ch];
            y = (y > 0.0f) ? y : 0.2f * y;
            hT[pbase + ch] = (__bf16)y;
        }
    }
}

// ---------------------------------------------------------------------------
// conv2: 3x3, 128->3 (rows padded to 16), reflect pad, sigmoid.
// Block = 128-pixel strip, one wave per 16-pixel tile. Hidden patch
// (3 x 130 x 128 bf16 = 99840 B) staged in dynamic LDS (TDM when interior).
// K = 9 taps x 4 chunks -> 36 WMMAs per wave, dual accumulators.
// ---------------------------------------------------------------------------
__global__ void __launch_bounds__(256)
conv2_wmma_kernel(const __bf16* __restrict__ hT,
                  const __bf16* __restrict__ wB,
                  const float*  __restrict__ b2,
                  float* __restrict__ out1) {
    extern __shared__ __bf16 sh[];                // 3*130*128 bf16 = 99840 B
    int wrow = blockIdx.x >> 2;
    int col0 = (blockIdx.x & 3) << 7;
    int wave = threadIdx.x >> 5;
    int lane = threadIdx.x & 31;
    int n = lane & 15;
    int cbase = (lane < 16) ? 0 : 16;

    bool interior = (wrow > 0) && (wrow < 511) && (col0 > 0) && (col0 < 384);
#if HAVE_TDM
    if (interior) {
        if (threadIdx.x < 32) {
            unsigned long long gaddr = (unsigned long long)(uintptr_t)hT +
                (((unsigned long long)((wrow - 1) * WDIM + col0 - 1)) << 8);
            unsigned lds_off = (unsigned)(uintptr_t)&sh[0];
            // 3 rows x (130*128*2/8 = 4160) 8B elems, row pitch 512*128*2/8 = 16384
            tdm_load_2d(gaddr, lds_off, 4160u, 3u, 16384ull);
            __builtin_amdgcn_s_wait_tensorcnt((short)0);
        }
        __syncthreads();
    } else
#endif
    {
        for (int t = threadIdx.x; t < 3 * 130; t += 256) {
            int r = t / 130, c = t % 130;
            int rr = refl(wrow + r - 1);
            int cc = refl(col0 + c - 1);
            const uint4* src = (const uint4*)(hT + (((size_t)(rr * WDIM + cc)) << 7));
            uint4* dst = (uint4*)(sh + t * 128);
#pragma unroll
            for (int q = 0; q < 16; ++q) dst[q] = src[q];
        }
        __syncthreads();
    }

    v8f acc0 = {}, acc1 = {};
#pragma unroll
    for (int i = 0; i < 3; ++i) {
#pragma unroll
        for (int j = 0; j < 3; ++j) {
            const __bf16* bp = sh + (((i * 130 + wave * 16 + n + j)) << 7) + cbase;
#pragma unroll
            for (int q = 0; q < 4; ++q) {
                v16bf a = *(const v16bf*)(wB + ((((i * 3 + j) * 4 + q) << 9) + (lane << 4)));
                v16bf b = *(const v16bf*)(bp + (q << 5));
                if (q & 1)
                    acc1 = __builtin_amdgcn_wmma_f32_16x16x32_bf16(
                        false, a, false, b, (short)0, acc1, false, false);
                else
                    acc0 = __builtin_amdgcn_wmma_f32_16x16x32_bf16(
                        false, a, false, b, (short)0, acc0, false, false);
            }
        }
    }

    // rows 0..2 live in acc[0..2] of lanes 0..15
    if (lane < 16) {
        int p = wrow * WDIM + col0 + wave * 16 + lane;
#pragma unroll
        for (int c = 0; c < 3; ++c) {
            float y = acc0[c] + acc1[c] + b2[c];
            out1[c * NPIX + p] = 1.0f / (1.0f + __expf(-y));
        }
    }
}

// ---------------------------------------------------------------------------
// Per-pixel tail: output2/3/4 adds, 2x2 & 3x3 symmetric expm, gradients,
// connection terms, quadratic forms, sqrt norm.
// ---------------------------------------------------------------------------
__device__ __forceinline__ void mm3(const float* A, const float* B, float* C) {
#pragma unroll
    for (int i = 0; i < 3; ++i)
#pragma unroll
        for (int j = 0; j < 3; ++j)
            C[i * 3 + j] = A[i * 3 + 0] * B[0 * 3 + j]
                         + A[i * 3 + 1] * B[1 * 3 + j]
                         + A[i * 3 + 2] * B[2 * 3 + j];
}

// s6 = [s00, s01, s02, s11, s12, s22]
__device__ void exp_sym3(const float* s6, float* E) {
    float A[9] = { s6[0], s6[1], s6[2],
                   s6[1], s6[3], s6[4],
                   s6[2], s6[4], s6[5] };
    float q  = (A[0] + A[4] + A[8]) * (1.0f / 3.0f);
    float p1 = A[1] * A[1] + A[2] * A[2] + A[5] * A[5];
    float b0 = A[0] - q, b4 = A[4] - q, b8 = A[8] - q;
    float p2 = b0 * b0 + b4 * b4 + b8 * b8 + 2.0f * p1;
    float pm = sqrtf(p2 * (1.0f / 6.0f));
    float l0, l1, l2;
    if (pm < 1e-10f) {
        l0 = l1 = l2 = q;
    } else {
        float inv = 1.0f / pm;
        float B0 = b0 * inv, B4 = b4 * inv, B8 = b8 * inv;
        float B1 = A[1] * inv, B2 = A[2] * inv, B5 = A[5] * inv;
        float detB = B0 * (B4 * B8 - B5 * B5)
                   - B1 * (B1 * B8 - B5 * B2)
                   + B2 * (B1 * B5 - B4 * B2);
        float r = fminf(1.0f, fmaxf(-1.0f, 0.5f * detB));
        float phi = acosf(r) * (1.0f / 3.0f);
        l0 = q + 2.0f * pm * __cosf(phi);
        l2 = q + 2.0f * pm * __cosf(phi + 2.0943951024f);
        l1 = 3.0f * q - l0 - l2;
    }
    float mx = fmaxf(fabsf(l0), fmaxf(fabsf(l1), fabsf(l2)));
    if (mx < 0.01f) {                       // "small" -> identity
        E[0] = 1.f; E[1] = 0.f; E[2] = 0.f;
        E[3] = 0.f; E[4] = 1.f; E[5] = 0.f;
        E[6] = 0.f; E[7] = 0.f; E[8] = 1.f;
        return;
    }
    if (pm < 1e-6f) {                       // A ~ q*I
        float e = __expf(q);
        E[0] = e; E[1] = 0.f; E[2] = 0.f;
        E[3] = 0.f; E[4] = e; E[5] = 0.f;
        E[6] = 0.f; E[7] = 0.f; E[8] = e;
        return;
    }
    // Sylvester: E = sum_i e^{li} (A-lj)(A-lk)/((li-lj)(li-lk))
    float lam[3] = { l0, l1, l2 };
#pragma unroll
    for (int t = 0; t < 9; ++t) E[t] = 0.0f;
#pragma unroll
    for (int i = 0; i < 3; ++i) {
        int j = (i + 1) % 3, k = (i + 2) % 3;
        float M1[9], M2[9], P[9];
#pragma unroll
        for (int t = 0; t < 9; ++t) { M1[t] = A[t]; M2[t] = A[t]; }
        M1[0] -= lam[j]; M1[4] -= lam[j]; M1[8] -= lam[j];
        M2[0] -= lam[k]; M2[4] -= lam[k]; M2[8] -= lam[k];
        mm3(M1, M2, P);
        float den = (lam[i] - lam[j]) * (lam[i] - lam[k]);
        if (fabsf(den) < 1e-12f) den = (den < 0.0f) ? -1e-12f : 1e-12f;
        float c = __expf(lam[i]) / den;
#pragma unroll
        for (int t = 0; t < 9; ++t) E[t] += c * P[t];
    }
}

__global__ void __launch_bounds__(256)
post_kernel(const float* __restrict__ x,
            const float* __restrict__ pr2,
            const float* __restrict__ pr3,
            const float* __restrict__ pr4,
            float* __restrict__ out) {
    int p = blockIdx.x * blockDim.x + threadIdx.x;
    if (p >= NPIX) return;
    int w = p >> 9, h = p & 511;

    const float* x2 = x + 32 * NPIX;
    const float* x3 = x + 35 * NPIX;
    const float* x4 = x + 41 * NPIX;
    float* o1 = out;
    float* o2 = out + 3 * NPIX;
    float* o3 = out + 6 * NPIX;
    float* o4 = out + 12 * NPIX;
    float* nrm = out + 30 * NPIX;

    // ---- output2 + 2x2 symmetric expm -> g
    float a2 = x2[p] + pr2[p];
    float bb = x2[NPIX + p] + pr2[NPIX + p];
    float c2 = x2[2 * NPIX + p] + pr2[2 * NPIX + p];
    o2[p] = a2; o2[NPIX + p] = bb; o2[2 * NPIX + p] = c2;
    float tmean = 0.5f * (a2 + c2);
    float dl = 0.5f * (a2 - c2);
    float s = sqrtf(dl * dl + bb * bb);
    float lA = tmean + s, lB = tmean - s;
    float g00, g01, g11;
    if (fmaxf(fabsf(lA), fabsf(lB)) < 0.01f) {
        g00 = 1.0f; g01 = 0.0f; g11 = 1.0f;
    } else {
        float et = __expf(tmean);
        float ch = coshf(s);
        float shs = (s > 1e-12f) ? (sinhf(s) / s) : 1.0f;
        g00 = et * (ch + shs * dl);
        g01 = et * (shs * bb);
        g11 = et * (ch - shs * dl);
    }

    // ---- output3 + 3x3 symmetric expm -> E
    float s6[6];
#pragma unroll
    for (int c = 0; c < 6; ++c) {
        s6[c] = x3[c * NPIX + p] + pr3[c * NPIX + p];
        o3[c * NPIX + p] = s6[c];
    }
    float E[9];
    exp_sym3(s6, E);

    // ---- output4 + connections
    float cx[9], cy[9];
#pragma unroll
    for (int k = 0; k < 9; ++k) {
        cx[k] = x4[k * NPIX + p] + pr4[k * NPIX + p];
        o4[k * NPIX + p] = cx[k];
    }
#pragma unroll
    for (int k = 0; k < 9; ++k) {
        cy[k] = x4[(9 + k) * NPIX + p] + pr4[(9 + k) * NPIX + p];
        o4[(9 + k) * NPIX + p] = cy[k];
    }

    // ---- output1 values + edge-padded central gradients
    int hp = (h < 511) ? h + 1 : 511, hm = (h > 0) ? h - 1 : 0;
    int wp = (w < 511) ? w + 1 : 511, wm = (w > 0) ? w - 1 : 0;
    float o1v[3], dx[3], dy[3];
#pragma unroll
    for (int c = 0; c < 3; ++c) {
        const float* b = o1 + c * NPIX;
        o1v[c] = b[p];
        dx[c] = 0.5f * (b[w * WDIM + hp] - b[w * WDIM + hm]);
        dy[c] = 0.5f * (b[wp * WDIM + h] - b[wm * WDIM + h]);
    }

    float cdx[3], cdy[3];
#pragma unroll
    for (int i = 0; i < 3; ++i) {
        cdx[i] = dx[i] + cx[i * 3] * o1v[0] + cx[i * 3 + 1] * o1v[1] + cx[i * 3 + 2] * o1v[2];
        cdy[i] = dy[i] + cy[i * 3] * o1v[0] + cy[i * 3 + 1] * o1v[1] + cy[i * 3 + 2] * o1v[2];
    }

    float Ex[3], Ey[3];
#pragma unroll
    for (int i = 0; i < 3; ++i) {
        Ex[i] = E[i * 3] * cdx[0] + E[i * 3 + 1] * cdx[1] + E[i * 3 + 2] * cdx[2];
        Ey[i] = E[i * 3] * cdy[0] + E[i * 3 + 1] * cdy[1] + E[i * 3 + 2] * cdy[2];
    }
    float qxx = cdx[0] * Ex[0] + cdx[1] * Ex[1] + cdx[2] * Ex[2];
    float qxy = cdx[0] * Ey[0] + cdx[1] * Ey[1] + cdx[2] * Ey[2];
    float qyy = cdy[0] * Ey[0] + cdy[1] * Ey[1] + cdy[2] * Ey[2];

    float tmp = g00 * qxx + 2.0f * g01 * qxy + g11 * qyy;
    nrm[h * WDIM + w] = sqrtf(tmp);          // flat order: n = h*W + w
}

// ---------------------------------------------------------------------------
// Launch
// ---------------------------------------------------------------------------
extern "C" void kernel_launch(void* const* d_in, const int* in_sizes, int n_in,
                              void* d_out, int out_size, void* d_ws, size_t ws_size,
                              hipStream_t stream) {
    const float* x   = (const float*)d_in[0];
    const float* w1  = (const float*)d_in[1];
    const float* b1  = (const float*)d_in[2];
    const float* w2  = (const float*)d_in[3];
    const float* b2  = (const float*)d_in[4];
    const float* pr2 = (const float*)d_in[5];
    const float* pr3 = (const float*)d_in[6];
    const float* pr4 = (const float*)d_in[7];
    float* out = (float*)d_out;

    // workspace layout (bytes)
    const size_t OFF_WA  = 0;                              // 8*9*512 bf16   = 73728 B
    const size_t OFF_WB  = 73728;                          // 36*512 bf16    = 36864 B
    const size_t OFF_X1T = 110592;                         // 262144*32 bf16 = 16 MiB
    const size_t OFF_HT  = 110592 + (size_t)16777216;      // 262144*128 bf16= 64 MiB
    char* ws = (char*)d_ws;
    __bf16* wA  = (__bf16*)(ws + OFF_WA);
    __bf16* wB  = (__bf16*)(ws + OFF_WB);
    __bf16* x1T = (__bf16*)(ws + OFF_X1T);
    __bf16* hT  = (__bf16*)(ws + OFF_HT);

    // 1) pack weights
    pack_weights_kernel<<<216, 256, 0, stream>>>(w1, w2, wA, wB);
    // 2) transpose + bf16-convert input (channel-innermost)
    pack_x1_kernel<<<NPIX / 256, 256, 0, stream>>>(x, x1T);
    // 3) conv1 via WMMA: 4096 blocks = 512 rows x 8 strips of 64 px
    conv1_wmma_kernel<<<4096, 256, 0, stream>>>(x1T, wA, b1, hT);
    // 4) conv2 via WMMA: 2048 blocks = 512 rows x 4 strips of 128 px,
    //    dynamic LDS = 3*130*128*2 = 99840 B (CDNA5: 320 KB/WGP available)
    conv2_wmma_kernel<<<2048, 256, 99840, stream>>>(hT, wB, b2, out);
    // 5) per-pixel tail math
    post_kernel<<<NPIX / 256, 256, 0, stream>>>(x, pr2, pr3, pr4, out);
}
